// HPBN_ADNI_60627758351012
// MI455X (gfx1250) — compile-verified, matrix operand order checked by reading, not run on hardware
//
#include <hip/hip_runtime.h>
#include <stdint.h>

typedef __attribute__((ext_vector_type(16))) __bf16 v16bf;
typedef __attribute__((ext_vector_type(8)))  float  v8f;

#define WMMA_BF16(a, b, c) \
    __builtin_amdgcn_wmma_f32_16x16x32_bf16(false, (a), false, (b), (short)0, (c), false, false)

// ---- sizes ----
#define BATCH   1024
#define NNODE   90
#define DIN     90
#define DN      48
#define KN      64
#define FD      8736
#define HID     256
#define OFF_NM0 0
#define OFF_NM1 4320
#define OFF_GP  8640
#define OFF_CP  8704

// ---- workspace layout (bytes) ----
#define WS_R    0                              // bf16 [1024][8736]   17,891,328 B
#define WS_MN   17891328                       // f32  [2][1024][48]     393,216 B
#define WS_H    (17891328 + 393216)            // f32  [1024][256]     1,048,576 B
#define WS_WT   (WS_H + 1048576)               // bf16 [256][8736]     4,472,832 B

// ---------------------------------------------------------------
// gfx1250 async global->LDS copy (16B per lane), tracked by ASYNCcnt.
// GV mode: lds byte-offset in vdst VGPR, 64-bit global address in vaddr pair.
// ---------------------------------------------------------------
__device__ inline void async_cp16(uint32_t lds_off, uint64_t gaddr) {
    asm volatile("global_load_async_to_lds_b128 %0, %1, off"
                 :: "v"(lds_off), "v"(gaddr) : "memory");
}
#define ASYNC_WAIT(N) asm volatile("s_wait_asynccnt %0" :: "i"(N) : "memory")

// Load a 16x32 bf16 A/B fragment from row-major LDS tile (tile base = row0,col0).
// Per ISA 7.12.2: lane m = lane&15; K runs [kb..kb+8) and [kb+16..kb+24), kb = kk + 8*lane[4].
__device__ inline v16bf ld_frag(const __bf16* tile, int ld, int kk) {
    const int lane = threadIdx.x & 31;
    const int m    = lane & 15;
    const int kb   = kk + ((lane >> 4) & 1) * 8;
    const __bf16* p = tile + m * ld + kb;
    v16bf r;
#pragma unroll
    for (int i = 0; i < 8; ++i) { r[i] = p[i]; r[8 + i] = p[16 + i]; }
    return r;
}

// =====================================================================
// Kernel 0: Wc1 [8736][256] f32 -> Wc1t [256][8736] bf16 (N-major),
// LDS-tiled 32x32 transpose, coalesced on both sides.
// =====================================================================
__global__ __launch_bounds__(256)
void k_cvt_wc1(const float* __restrict__ Wc1, __bf16* __restrict__ Wt) {
    __shared__ float tile[32][33];
    const int k0 = blockIdx.x * 32, n0 = blockIdx.y * 32;
#pragma unroll
    for (int i = 0; i < 4; ++i) {
        int kk = threadIdx.y + i * 8;
        tile[kk][threadIdx.x] = Wc1[(size_t)(k0 + kk) * HID + n0 + threadIdx.x];
    }
    __syncthreads();
#pragma unroll
    for (int i = 0; i < 4; ++i) {
        int nn = threadIdx.y + i * 8;
        Wt[(size_t)(n0 + nn) * FD + k0 + threadIdx.x] = (__bf16)tile[threadIdx.x][nn];
    }
}

// =====================================================================
// Kernel 1: node branch.  grid (1024, 2), block 192 (6 waves).
// =====================================================================
__global__ __launch_bounds__(192)
void k_node_branch(const float* __restrict__ fcn, const float* __restrict__ scn,
                   const float* __restrict__ W_node, const float* __restrict__ b_node,
                   const float* __restrict__ P_node,
                   __bf16* __restrict__ R, float* __restrict__ mn) {
    __shared__ __align__(16) char smem[57344];
    __shared__ float srn[96];
    __shared__ float srp[64];
    __shared__ float sMean[48];

    __bf16* sX  = (__bf16*)smem;              // [96][96]  18432 B (phase 1)
    __bf16* sWt = (__bf16*)(smem + 18432);    // [48][96]   9216 B (phase 1)
    __bf16* sNod = (__bf16*)smem;             // [96][64]  12288 B
    __bf16* sP   = (__bf16*)(smem + 12288);   // [64][64]   8192 B
    float*  sS   = (float* )(smem + 20480);   // [96][64]  24576 B
    __bf16* sWgt = (__bf16*)(smem + 45056);   // [96][64]  12288 B
    __bf16* sPT  = (__bf16*)smem;             // [48][64]   6144 B (reuses sNod)

    const int tid  = threadIdx.x;
    const int wave = tid >> 5;
    const int lane = tid & 31;
    const int nl   = lane & 15;
    const int g    = (lane >> 4) & 1;
    const int b    = blockIdx.x;
    const int br   = blockIdx.y;
    const float* X = (br ? scn : fcn) + (size_t)b * (DIN * NNODE);

    for (int idx = tid; idx < 96 * 96; idx += 192) {
        int m = idx / 96, k = idx - m * 96;
        sX[idx] = (m < NNODE && k < DIN) ? (__bf16)X[m * DIN + k] : (__bf16)0.f;
    }
    for (int idx = tid; idx < 48 * 96; idx += 192) {
        int n = idx / 96, k = idx - n * 96;
        sWt[idx] = (k < DIN) ? (__bf16)W_node[k * DN + n] : (__bf16)0.f;
    }
    __syncthreads();

    const int m0 = wave * 16;
    v8f accN[3] = {v8f(0.f), v8f(0.f), v8f(0.f)};
#pragma unroll
    for (int kk = 0; kk < 96; kk += 32) {
        v16bf a = ld_frag(sX + m0 * 96, 96, kk);
#pragma unroll
        for (int t = 0; t < 3; ++t) {
            v16bf bfr = ld_frag(sWt + t * 16 * 96, 96, kk);
            accN[t] = WMMA_BF16(a, bfr, accN[t]);
        }
    }
    __syncthreads();

    for (int idx = tid; idx < 96 * 64; idx += 192) sNod[idx] = (__bf16)0.f;
    for (int idx = tid; idx < 64 * 64; idx += 192) {
        int n = idx >> 6, k = idx & 63;
        sP[idx] = (k < DN) ? (__bf16)P_node[n * DN + k] : (__bf16)0.f;
    }
    if (tid < 64) {
        float s = 0.f;
        for (int k = 0; k < DN; ++k) { float v = P_node[tid * DN + k]; s += v * v; }
        srp[tid] = s;
    }
    __syncthreads();

#pragma unroll
    for (int t = 0; t < 3; ++t) {
        float bn = b_node[t * 16 + nl];
#pragma unroll
        for (int r = 0; r < 8; ++r) {
            int m = m0 + r + 8 * g;
            sNod[m * 64 + t * 16 + nl] = (__bf16)(accN[t][r] + bn);
        }
    }
    __syncthreads();

    if (tid < 96) {
        float s = 0.f;
        for (int k = 0; k < DN; ++k) { float v = (float)sNod[tid * 64 + k]; s += v * v; }
        srn[tid] = s;
    }
    __syncthreads();

    {
        v8f accS[4] = {v8f(0.f), v8f(0.f), v8f(0.f), v8f(0.f)};
#pragma unroll
        for (int kk = 0; kk < 64; kk += 32) {
            v16bf a = ld_frag(sNod + m0 * 64, 64, kk);
#pragma unroll
            for (int t = 0; t < 4; ++t) {
                v16bf bfr = ld_frag(sP + t * 16 * 64, 64, kk);
                accS[t] = WMMA_BF16(a, bfr, accS[t]);
            }
        }
#pragma unroll
        for (int t = 0; t < 4; ++t)
#pragma unroll
            for (int r = 0; r < 8; ++r) {
                int m = m0 + r + 8 * g;
                int n = t * 16 + nl;
                sS[m * 64 + n] = 2.f * accS[t][r] - srn[m] - srp[n];
            }
    }
    __syncthreads();

    for (int idx = tid; idx < 48 * 64; idx += 192) {
        int n = idx >> 6, k = idx & 63;
        sPT[idx] = (__bf16)P_node[k * DN + n];
    }
    if (tid < 48) sMean[tid] = 0.f;
    if (tid < 96) {
        float mx = -1e30f;
        for (int k = 0; k < KN; ++k) mx = fmaxf(mx, sS[tid * 64 + k]);
        float sum = 0.f;
        for (int k = 0; k < KN; ++k) {
            float e = __expf(sS[tid * 64 + k] - mx);
            sS[tid * 64 + k] = e; sum += e;
        }
        float inv = 1.f / sum;
        for (int k = 0; k < KN; ++k) sWgt[tid * 64 + k] = (__bf16)(sS[tid * 64 + k] * inv);
    }
    __syncthreads();

    v8f accM[3] = {v8f(0.f), v8f(0.f), v8f(0.f)};
#pragma unroll
    for (int kk = 0; kk < 64; kk += 32) {
        v16bf a = ld_frag(sWgt + m0 * 64, 64, kk);
#pragma unroll
        for (int t = 0; t < 3; ++t) {
            v16bf bfr = ld_frag(sPT + t * 16 * 64, 64, kk);
            accM[t] = WMMA_BF16(a, bfr, accM[t]);
        }
    }

    __bf16* Rrow = R + (size_t)b * FD + (br ? OFF_NM1 : OFF_NM0);
#pragma unroll
    for (int t = 0; t < 3; ++t)
#pragma unroll
        for (int r = 0; r < 8; ++r) {
            int m = m0 + r + 8 * g;
            if (m < NNODE) {
                int col = t * 16 + nl;
                float v = accM[t][r];
                atomicAdd(&sMean[col], v);
                Rrow[m * DN + col] = (__bf16)v;
            }
        }
    __syncthreads();
    if (tid < 48) mn[((size_t)br * BATCH + b) * DN + tid] = sMean[tid] * (1.f / 90.f);
}

// =====================================================================
// Kernel 2: graph path, one thread per (branch, sample).
// =====================================================================
__device__ inline void soft_proto(const float* e, const float* P, int K, int D, float* out, float* z) {
    float en = 0.f;
    for (int d = 0; d < D; ++d) en += e[d] * e[d];
    float mx = -1e30f;
    for (int k = 0; k < K; ++k) {
        float dot = 0.f, pn = 0.f;
        for (int d = 0; d < D; ++d) { float pv = P[k * D + d]; dot += e[d] * pv; pn += pv * pv; }
        z[k] = 2.f * dot - en - pn;
        mx = fmaxf(mx, z[k]);
    }
    float sum = 0.f;
    for (int k = 0; k < K; ++k) { z[k] = __expf(z[k] - mx); sum += z[k]; }
    float inv = 1.f / sum;
    for (int d = 0; d < D; ++d) out[d] = 0.f;
    for (int k = 0; k < K; ++k) {
        float w = z[k] * inv;
        for (int d = 0; d < D; ++d) out[d] += w * P[k * D + d];
    }
}

__global__ __launch_bounds__(128)
void k_graph_path(const float* __restrict__ mn,
                  const float* __restrict__ W_graph, const float* __restrict__ b_graph,
                  const float* __restrict__ P_graph,
                  const float* __restrict__ Wg1, const float* __restrict__ bg1,
                  const float* __restrict__ Wg2, const float* __restrict__ bg2,
                  const float* __restrict__ P_cls,
                  __bf16* __restrict__ R) {
    int tid = blockIdx.x * blockDim.x + threadIdx.x;
    if (tid >= 2 * BATCH) return;
    int br = tid >> 10, b = tid & (BATCH - 1);
    const float* m = mn + ((size_t)br * BATCH + b) * DN;

    float eg[32], gp[32], z[32];
    for (int j = 0; j < 32; ++j) {
        float a = b_graph[j];
        for (int k = 0; k < DN; ++k) a += m[k] * W_graph[k * 32 + j];
        eg[j] = a;
    }
    soft_proto(eg, P_graph, 32, 32, gp, z);
    __bf16* Rrow = R + (size_t)b * FD;
    for (int j = 0; j < 32; ++j) Rrow[OFF_GP + br * 32 + j] = (__bf16)gp[j];

    float h1[32];
    for (int j = 0; j < 32; ++j) {
        float a = bg1[j];
        for (int k = 0; k < 32; ++k) a += gp[k] * Wg1[k * 32 + j];
        h1[j] = fmaxf(a, 0.f);
    }
    float ec[16], cp[16];
    for (int j = 0; j < 16; ++j) {
        float a = bg2[j];
        for (int k = 0; k < 32; ++k) a += h1[k] * Wg2[k * 16 + j];
        ec[j] = a;
    }
    soft_proto(ec, P_cls, 16, 16, cp, z);
    for (int j = 0; j < 16; ++j) Rrow[OFF_CP + br * 16 + j] = (__bf16)cp[j];
}

// =====================================================================
// Kernel 3: H = relu(R @ Wc1 + bc1).  M=1024 K=8736 N=256.
// grid (16,8), block 128 (4 waves). Block tile 64x32, K chunk 96 (96*91=8736).
// Double-buffered LDS filled by GLOBAL_LOAD_ASYNC_TO_LDS_B128; each wave
// issues 9 async ops per chunk, so s_wait_asynccnt 9 drains the previous
// chunk (async loads complete in order) while the next chunk is in flight.
// =====================================================================
__global__ __launch_bounds__(128)
void k_classifier(const __bf16* __restrict__ R, const __bf16* __restrict__ Wt,
                  const float* __restrict__ bc1, float* __restrict__ H) {
    __shared__ __align__(16) __bf16 sA[2][64 * 96];   // 2 x 12288 B
    __shared__ __align__(16) __bf16 sB[2][32 * 96];   // 2 x  6144 B

    const int tid  = threadIdx.x;
    const int wave = tid >> 5;
    const int lane = tid & 31;
    const int nl   = lane & 15;
    const int g    = (lane >> 4) & 1;
    const int mB   = blockIdx.x * 64;
    const int nB   = blockIdx.y * 32;

    // per-thread copy assignments (16B units)
    const int arow = tid >> 1, ahalf = (tid & 1) * 96;   // 2 thr/row, 96 B each (6 units)
    const int brow = tid >> 2, bq    = (tid & 3) * 48;   // 4 thr/row, 48 B each (3 units)
    const uint64_t Rbase = (uint64_t)(uintptr_t)R + (uint64_t)(mB + arow) * (FD * 2) + ahalf;
    const uint64_t Wbase = (uint64_t)(uintptr_t)Wt + (uint64_t)(nB + brow) * (FD * 2) + bq;
    const uint32_t sAoff0 = (uint32_t)(uintptr_t)&sA[0][0] + arow * 192 + ahalf;
    const uint32_t sAoff1 = (uint32_t)(uintptr_t)&sA[1][0] + arow * 192 + ahalf;
    const uint32_t sBoff0 = (uint32_t)(uintptr_t)&sB[0][0] + brow * 192 + bq;
    const uint32_t sBoff1 = (uint32_t)(uintptr_t)&sB[1][0] + brow * 192 + bq;

    v8f acc[2] = {v8f(0.f), v8f(0.f)};

    // issue chunk kb into buffer kb&1 (9 async instructions per thread/wave)
    auto issue = [&](int kb) {
        const uint64_t gA = Rbase + (uint64_t)kb * 192;
        const uint64_t gB = Wbase + (uint64_t)kb * 192;
        const uint32_t lA = (kb & 1) ? sAoff1 : sAoff0;
        const uint32_t lB = (kb & 1) ? sBoff1 : sBoff0;
#pragma unroll
        for (int i = 0; i < 6; ++i) async_cp16(lA + i * 16, gA + i * 16);
#pragma unroll
        for (int i = 0; i < 3; ++i) async_cp16(lB + i * 16, gB + i * 16);
    };

    issue(0);
    for (int kb = 0; kb < 91; ++kb) {
        if (kb + 1 < 91) { issue(kb + 1); ASYNC_WAIT(9); }
        else             { ASYNC_WAIT(0); }
        __syncthreads();   // chunk kb resident in LDS for all waves

        const __bf16* cA = sA[kb & 1];
        const __bf16* cB = sB[kb & 1];
#pragma unroll
        for (int kk = 0; kk < 96; kk += 32) {
            v16bf a = ld_frag(cA + wave * 16 * 96, 96, kk);
#pragma unroll
            for (int t = 0; t < 2; ++t) {
                v16bf bfr = ld_frag(cB + t * 16 * 96, 96, kk);
                acc[t] = WMMA_BF16(a, bfr, acc[t]);
            }
        }
        __syncthreads();   // all waves done reading before buffer reuse
    }

#pragma unroll
    for (int t = 0; t < 2; ++t) {
        int gn = nB + t * 16 + nl;
        float bias = bc1[gn];
#pragma unroll
        for (int r = 0; r < 8; ++r) {
            int gm = mB + wave * 16 + r + 8 * g;
            H[(size_t)gm * HID + gn] = fmaxf(acc[t][r] + bias, 0.f);
        }
    }
}

// =====================================================================
// Kernel 4: logits = H @ Wc2 + bc2   (N=3)
// =====================================================================
__global__ __launch_bounds__(128)
void k_logits(const float* __restrict__ H, const float* __restrict__ Wc2,
              const float* __restrict__ bc2, float* __restrict__ out) {
    int i = blockIdx.x * blockDim.x + threadIdx.x;
    if (i >= BATCH * 3) return;
    int row = i / 3, col = i - row * 3;
    float s = bc2[col];
    const float* h = H + (size_t)row * HID;
    for (int k = 0; k < HID; ++k) s += h[k] * Wc2[k * 3 + col];
    out[i] = s;
}

extern "C" void kernel_launch(void* const* d_in, const int* in_sizes, int n_in,
                              void* d_out, int out_size, void* d_ws, size_t ws_size,
                              hipStream_t stream) {
    (void)in_sizes; (void)n_in; (void)out_size; (void)ws_size;
    const float* fcn     = (const float*)d_in[0];
    const float* scn     = (const float*)d_in[1];
    const float* W_node  = (const float*)d_in[2];
    const float* b_node  = (const float*)d_in[3];
    const float* P_node  = (const float*)d_in[4];
    const float* W_graph = (const float*)d_in[5];
    const float* b_graph = (const float*)d_in[6];
    const float* P_graph = (const float*)d_in[7];
    const float* Wg1     = (const float*)d_in[8];
    const float* bg1     = (const float*)d_in[9];
    const float* Wg2     = (const float*)d_in[10];
    const float* bg2     = (const float*)d_in[11];
    const float* P_cls   = (const float*)d_in[12];
    const float* Wc1     = (const float*)d_in[13];
    const float* bc1     = (const float*)d_in[14];
    const float* Wc2     = (const float*)d_in[15];
    const float* bc2     = (const float*)d_in[16];

    char* ws = (char*)d_ws;
    __bf16* R   = (__bf16*)(ws + WS_R);
    float*  mn  = (float* )(ws + WS_MN);
    float*  H   = (float* )(ws + WS_H);
    __bf16* Wt  = (__bf16*)(ws + WS_WT);
    float*  out = (float*)d_out;

    k_cvt_wc1<<<dim3(FD / 32, HID / 32), dim3(32, 8), 0, stream>>>(Wc1, Wt);
    k_node_branch<<<dim3(BATCH, 2), 192, 0, stream>>>(fcn, scn, W_node, b_node, P_node, R, mn);
    k_graph_path<<<(2 * BATCH + 127) / 128, 128, 0, stream>>>(
        mn, W_graph, b_graph, P_graph, Wg1, bg1, Wg2, bg2, P_cls, R);
    k_classifier<<<dim3(BATCH / 64, HID / 32), 128, 0, stream>>>(R, Wt, bc1, H);
    k_logits<<<(BATCH * 3 + 127) / 128, 128, 0, stream>>>(H, Wc2, bc2, out);
}
